// UPerNet_Sphere_82506321756170
// MI455X (gfx1250) — matrix operand order repaired and checked by reading, not compiled, not run on hardware
//
#include <hip/hip_runtime.h>
#include <cfloat>

typedef __attribute__((ext_vector_type(16))) __bf16 v16bf;
typedef __attribute__((ext_vector_type(8)))  float  v8f;

#define CDIV(a,b) (((a)+(b)-1)/(b))

// =====================================================================
// Implicit-GEMM conv via V_WMMA_F32_16X16X32_BF16 (wave32).
//   out[b][co][p] = epilogue( sum_{ci,r,c} W[co][ci*KS*KS+r*KS+c] *
//                             in[b][ci][y+r-PAD][x+c-PAD] )
// Block: 256 threads (8 waves). Block tile: 64 (cout) x 128 (pixels).
// Wave w owns N-subtile w (16 pixels) and all 4 M-subtiles -> 4 WMMAs
// per K-chunk sharing one B fragment. Double-buffered LDS staging.
// MODE: 0 = raw, 1 = relu(scale*x + bias), 2 = x + bias
// =====================================================================
template<int KS, int MODE>
__global__ __launch_bounds__(256) void conv_wmma_kernel(
    const float* __restrict__ in, long in_bs, int Cin, int H, int W,
    const float* __restrict__ Wgt, const float* __restrict__ Sc,
    const float* __restrict__ Bi,
    float* __restrict__ out, long out_bs, int Cout)
{
    const int HW = H * W;
    const int K  = Cin * KS * KS;          // multiple of 32 at all call sites
    const int nBase = blockIdx.x * 128;    // pixel tile
    const int mBase = blockIdx.y * 64;     // cout tile
    const int b     = blockIdx.z;

    __shared__ __bf16 lA[2][64][34];       // A[m][k]  (weights), padded stride
    __shared__ __bf16 lB[2][128][34];      // B^T[n][k] (im2col), padded stride

    const int t    = threadIdx.x;
    const int lane = t & 31;
    const int wn   = t >> 5;               // wave id == N subtile (16 cols)
    const int lm   = lane & 15;
    const int hi   = lane >> 4;            // 0 or 1

    const float* inB = in + (long)b * in_bs;

    // ---- staging helpers (fp32 -> bf16 into LDS) ----
    auto stageA = [&](int buf, int k0) {
        #pragma unroll
        for (int i = 0; i < 8; ++i) {      // 64x32 = 2048 elems, 8/thread
            int idx = t + i * 256;
            int m = idx >> 5, kk = idx & 31;
            int gm = mBase + m;
            float v = (gm < Cout) ? Wgt[(long)gm * K + (k0 + kk)] : 0.0f;
            lA[buf][m][kk] = (__bf16)v;
        }
    };
    auto stageB = [&](int buf, int k0) {
        #pragma unroll
        for (int i = 0; i < 16; ++i) {     // 128x32 = 4096 elems, 16/thread
            int idx = t + i * 256;
            int n = idx >> 5, kk = idx & 31;
            int gk = k0 + kk;
            int p  = nBase + n;
            float v;
            if (KS == 1) {
                v = inB[(long)gk * HW + p];
                if (gk + 64 < K)           // GL2 prefetch, 2 chunks ahead
                    __builtin_prefetch(&inB[(long)(gk + 64) * HW + p], 0, 1);
            } else {
                int ci = gk / 9, rem = gk % 9;
                int r = rem / 3, c = rem % 3;
                int y = p / W, x = p % W;
                int iy = y + r - 1, ix = x + c - 1;
                v = (iy >= 0 && iy < H && ix >= 0 && ix < W)
                    ? inB[(long)ci * HW + (long)iy * W + ix]
                    : 0.0f;
            }
            lB[buf][n][kk] = (__bf16)v;
        }
    };

    v8f acc0 = {}, acc1 = {}, acc2 = {}, acc3 = {};

    stageA(0, 0);
    stageB(0, 0);

    int buf = 0;
    for (int k0 = 0; k0 < K; k0 += 32) {
        __syncthreads();                   // staging of `buf` complete;
                                           // all reads of buf^1 finished
        if (k0 + 32 < K) {                 // overlap next-chunk staging
            stageA(buf ^ 1, k0 + 32);
            stageB(buf ^ 1, k0 + 32);
        }

        // ---- fragments per ISA VGPR layouts (wave32) ----
        v16bf a0, a1, a2, a3, bb;
        const int nrow = wn * 16 + lm;     // B column
        const int kB   = hi << 4;          // B: same k pattern as A (kk)
        #pragma unroll
        for (int i = 0; i < 16; ++i) {
            int kk = ((i >> 3) << 4) + (hi << 3) + (i & 7);  // A k pattern
            bb[i] = lB[buf][nrow][kk];
            a0[i] = lA[buf][lm][kk];
            a1[i] = lA[buf][16 + lm][kk];
            a2[i] = lA[buf][32 + lm][kk];
            a3[i] = lA[buf][48 + lm][kk];
        }
        acc0 = __builtin_amdgcn_wmma_f32_16x16x32_bf16(
                   false, a0, false, bb, (short)0, acc0, false, false);
        acc1 = __builtin_amdgcn_wmma_f32_16x16x32_bf16(
                   false, a1, false, bb, (short)0, acc1, false, false);
        acc2 = __builtin_amdgcn_wmma_f32_16x16x32_bf16(
                   false, a2, false, bb, (short)0, acc2, false, false);
        acc3 = __builtin_amdgcn_wmma_f32_16x16x32_bf16(
                   false, a3, false, bb, (short)0, acc3, false, false);
        buf ^= 1;
    }

    // ---- epilogue: D layout row = r + 8*(lane/16), col = lane%16 ----
    const int pcol = nBase + wn * 16 + lm;
    const int rb   = hi << 3;
    float* outB = out + (long)b * out_bs;
    #pragma unroll
    for (int r = 0; r < 8; ++r) {
        #pragma unroll
        for (int s = 0; s < 4; ++s) {
            int co = mBase + s * 16 + rb + r;
            if (co < Cout) {
                float v = (s == 0) ? acc0[r] : (s == 1) ? acc1[r]
                        : (s == 2) ? acc2[r] : acc3[r];
                if constexpr (MODE == 1) v = fmaxf(v * Sc[co] + Bi[co], 0.0f);
                if constexpr (MODE == 2) v = v + Bi[co];
                outB[(long)co * HW + pcol] = v;
            }
        }
    }
}

// =====================================================================
// Bilinear resize (half-pixel, align_corners=False) with optional fused
// residual add. Also serves as a strided channel-slice copy (Hin==Hout).
// =====================================================================
__global__ void resize_bilinear_kernel(
    const float* __restrict__ in, long in_bs, int C, int Hin, int Win,
    float* __restrict__ out, long out_bs, int Hout, int Wout,
    const float* __restrict__ add, long add_bs, long total)
{
    long idx = (long)blockIdx.x * blockDim.x + threadIdx.x;
    if (idx >= total) return;
    int x = (int)(idx % Wout);
    long tq = idx / Wout;
    int y = (int)(tq % Hout); tq /= Hout;
    int c = (int)(tq % C);
    int b = (int)(tq / C);

    float sy = (y + 0.5f) * (float)Hin / (float)Hout - 0.5f;
    float sx = (x + 0.5f) * (float)Win / (float)Wout - 0.5f;
    int y0 = (int)floorf(sy); float wy = sy - (float)y0;
    int x0 = (int)floorf(sx); float wx = sx - (float)x0;
    int y0c = min(max(y0, 0), Hin - 1), y1c = min(max(y0 + 1, 0), Hin - 1);
    int x0c = min(max(x0, 0), Win - 1), x1c = min(max(x0 + 1, 0), Win - 1);

    const float* p = in + (long)b * in_bs + (long)c * Hin * Win;
    float v00 = p[(long)y0c * Win + x0c], v01 = p[(long)y0c * Win + x1c];
    float v10 = p[(long)y1c * Win + x0c], v11 = p[(long)y1c * Win + x1c];
    float v = (1.0f - wy) * ((1.0f - wx) * v00 + wx * v01)
            +          wy * ((1.0f - wx) * v10 + wx * v11);
    long opix = (long)c * Hout * Wout + (long)y * Wout + x;
    if (add) v += add[(long)b * add_bs + opix];
    out[(long)b * out_bs + opix] = v;
}

// exact torch AdaptiveAvgPool2d (floor start / ceil end bins)
__global__ void adaptive_pool_kernel(
    const float* __restrict__ in, int C, int H, int W, int s,
    float* __restrict__ out, long total)
{
    long idx = (long)blockIdx.x * blockDim.x + threadIdx.x;
    if (idx >= total) return;
    int j = (int)(idx % s);
    long tq = idx / s;
    int i = (int)(tq % s); tq /= s;
    int c = (int)(tq % C);
    int b = (int)(tq / C);
    int h0 = (i * H) / s, h1 = ((i + 1) * H + s - 1) / s;
    int w0 = (j * W) / s, w1 = ((j + 1) * W + s - 1) / s;
    const float* p = in + ((long)b * C + c) * H * W;
    float sum = 0.0f;
    for (int y = h0; y < h1; ++y)
        for (int x = w0; x < w1; ++x) sum += p[(long)y * W + x];
    out[idx] = sum / (float)((h1 - h0) * (w1 - w0));
}

__global__ void maxpool4_kernel(
    const float* __restrict__ in, int C, int H, int W,
    float* __restrict__ out, long total)
{
    long idx = (long)blockIdx.x * blockDim.x + threadIdx.x;
    if (idx >= total) return;
    int Wo = W / 4, Ho = H / 4;
    int x = (int)(idx % Wo);
    long tq = idx / Wo;
    int y = (int)(tq % Ho); tq /= Ho;
    int c = (int)(tq % C);
    int b = (int)(tq / C);
    const float* p = in + ((long)b * C + c) * H * W;
    float m = -FLT_MAX;
    #pragma unroll
    for (int dy = 0; dy < 4; ++dy)
        #pragma unroll
        for (int dx = 0; dx < 4; ++dx)
            m = fmaxf(m, p[(long)(y * 4 + dy) * W + (x * 4 + dx)]);
    out[idx] = m;
}

// wnT[k][c] = w[c][k] / ||w[:,k]||   (w is [Cin][K] row-major)
__global__ void normalize_fc_kernel(
    const float* __restrict__ w, float* __restrict__ wnT, int Cin, int K)
{
    int k = blockIdx.x * blockDim.x + threadIdx.x;
    if (k >= K) return;
    float ss = 0.0f;
    for (int c = 0; c < Cin; ++c) { float v = w[(long)c * K + k]; ss += v * v; }
    float r = 1.0f / sqrtf(ss);
    for (int c = 0; c < Cin; ++c) wnT[(long)k * Cin + c] = w[(long)c * K + k] * r;
}

// per-pixel KL(softmax(xe)||softmax(xs)) over C channels; score = 1 - tanh(kl)
__global__ void kl_score_kernel(
    const float* __restrict__ xe, const float* __restrict__ xs,
    float* __restrict__ sc, int C, int HW, long total)
{
    long idx = (long)blockIdx.x * blockDim.x + threadIdx.x;
    if (idx >= total) return;
    long b = idx / HW, p = idx % HW;
    const float* pe = xe + (long)b * C * HW + p;
    const float* ps = xs + (long)b * C * HW + p;
    float me = -FLT_MAX, ms = -FLT_MAX;
    for (int c = 0; c < C; ++c) {
        me = fmaxf(me, pe[(long)c * HW]);
        ms = fmaxf(ms, ps[(long)c * HW]);
    }
    float se = 0.0f, ss = 0.0f;
    for (int c = 0; c < C; ++c) {
        se += expf(pe[(long)c * HW] - me);
        ss += expf(ps[(long)c * HW] - ms);
    }
    float lse = logf(se), lss = logf(ss);
    float kl = 0.0f;
    for (int c = 0; c < C; ++c) {
        float lpe = pe[(long)c * HW] - me - lse;
        float lps = ps[(long)c * HW] - ms - lss;
        kl += expf(lpe) * (lpe - lps);
    }
    sc[idx] = 1.0f - tanhf(kl);
}

// fused bilinear upsample (128^2 -> 512^2) + channel softmax -> d_out
__global__ void resize_softmax_kernel(
    const float* __restrict__ xin, int C, int Hin, int Win,
    float* __restrict__ outp, int Hout, int Wout, long total)
{
    long idx = (long)blockIdx.x * blockDim.x + threadIdx.x;
    if (idx >= total) return;
    int x = (int)(idx % Wout);
    long tq = idx / Wout;
    int y = (int)(tq % Hout);
    int b = (int)(tq / Hout);

    float sy = (y + 0.5f) * (float)Hin / (float)Hout - 0.5f;
    float sx = (x + 0.5f) * (float)Win / (float)Wout - 0.5f;
    int y0 = (int)floorf(sy); float wy = sy - (float)y0;
    int x0 = (int)floorf(sx); float wx = sx - (float)x0;
    int y0c = min(max(y0, 0), Hin - 1), y1c = min(max(y0 + 1, 0), Hin - 1);
    int x0c = min(max(x0, 0), Win - 1), x1c = min(max(x0 + 1, 0), Win - 1);
    float w00 = (1.0f - wy) * (1.0f - wx), w01 = (1.0f - wy) * wx;
    float w10 = wy * (1.0f - wx),          w11 = wy * wx;
    long i00 = (long)y0c * Win + x0c, i01 = (long)y0c * Win + x1c;
    long i10 = (long)y1c * Win + x0c, i11 = (long)y1c * Win + x1c;

    const long HWin = (long)Hin * Win, HWout = (long)Hout * Wout;
    const float* base = xin + (long)b * C * HWin;
    float m = -FLT_MAX;
    for (int c = 0; c < C; ++c) {
        const float* p = base + (long)c * HWin;
        float v = w00 * p[i00] + w01 * p[i01] + w10 * p[i10] + w11 * p[i11];
        m = fmaxf(m, v);
    }
    float s = 0.0f;
    for (int c = 0; c < C; ++c) {
        const float* p = base + (long)c * HWin;
        float v = w00 * p[i00] + w01 * p[i01] + w10 * p[i10] + w11 * p[i11];
        s += expf(v - m);
    }
    float inv = 1.0f / s;
    for (int c = 0; c < C; ++c) {
        const float* p = base + (long)c * HWin;
        float v = w00 * p[i00] + w01 * p[i01] + w10 * p[i10] + w11 * p[i11];
        outp[((long)b * C + c) * HWout + (long)y * Wout + x] = expf(v - m) * inv;
    }
}

// =====================================================================
extern "C" void kernel_launch(void* const* d_in, const int* in_sizes, int n_in,
                              void* d_out, int out_size, void* d_ws, size_t ws_size,
                              hipStream_t stream)
{
    (void)in_sizes; (void)n_in; (void)out_size; (void)ws_size;
    const int B = 2;
    const long HW16 = 256, HW32 = 1024, HW64 = 4096, HW128 = 16384, HW512 = 262144;

    // ---- inputs (depth-first dict insertion order) ----
    const float* conv0 = (const float*)d_in[0];   // [2,256,128,128]
    const float* conv1 = (const float*)d_in[1];   // [2,512,64,64]
    const float* conv2 = (const float*)d_in[2];   // [2,1024,32,32]
    const float* conv5 = (const float*)d_in[3];   // [2,2048,16,16]
    const float *ppm_w[4], *ppm_s[4], *ppm_b[4];
    for (int j = 0; j < 4; ++j) {
        ppm_w[j] = (const float*)d_in[4 + 3 * j];
        ppm_s[j] = (const float*)d_in[5 + 3 * j];
        ppm_b[j] = (const float*)d_in[6 + 3 * j];
    }
    const float* ppl_w = (const float*)d_in[16];
    const float* ppl_s = (const float*)d_in[17];
    const float* ppl_b = (const float*)d_in[18];
    const float *fin_w[3], *fin_s[3], *fin_b[3];
    for (int j = 0; j < 3; ++j) {
        fin_w[j] = (const float*)d_in[19 + 3 * j];
        fin_s[j] = (const float*)d_in[20 + 3 * j];
        fin_b[j] = (const float*)d_in[21 + 3 * j];
    }
    const float *fout_w[3], *fout_s[3], *fout_b[3];
    for (int j = 0; j < 3; ++j) {
        fout_w[j] = (const float*)d_in[28 + 3 * j];
        fout_s[j] = (const float*)d_in[29 + 3 * j];
        fout_b[j] = (const float*)d_in[30 + 3 * j];
    }
    const float* lc_w   = (const float*)d_in[37];
    const float* lc_s   = (const float*)d_in[38];
    const float* lc_b   = (const float*)d_in[39];
    const float* last_w = (const float*)d_in[40];  // [150,256]
    const float* last_b = (const float*)d_in[41];  // [150]
    const float* sp_w   = (const float*)d_in[42];
    const float* sp_s   = (const float*)d_in[43];
    const float* sp_b   = (const float*)d_in[44];
    const float* fcw    = (const float*)d_in[45];  // [256,150]
    float* out = (float*)d_out;
    float* ws  = (float*)d_ws;

    // ---- workspace layout (float offsets), with safe aliasing ----
    const long oFusion = 0;                       // [2,1024,128,128]
    const long oBufA   = 33554432;                // 8,388,608 floats
    const long oBufB   = oBufA + 8388608;
    const long oBufC   = oBufB + 8388608;         // x_e logits [2,150,128,128]
    const long oF128   = oBufC + 4915200;         // [2,256,128,128]
    const long oF16    = oF128 + 8388608;         // [2,256,16,16]
    const long oOut32  = oF16  + 131072;          // [2,256,32,32]
    const long oOut64  = oOut32 + 524288;         // [2,256,64,64]
    const long oMp     = oOut64 + 2097152;        // [2,256,32,32]
    const long oWnT    = oMp   + 524288;          // [150,256]
    const long oXs32   = oWnT  + 65536;           // [2,150,32,32]
    const long oScore  = oXs32 + 307200;          // [2,1,128,128]
    // aliases (lifetimes disjoint):
    const long oPpmCat  = oBufA;                  // [2,4096,16,16]
    const long oPoolRes = oBufA + 2097152;        // [2,2048,16,16]
    const long oPool    = oBufA + 3145728;        // up to [2,2048,6,6]
    const long oCx32 = oBufB, oF32 = oBufB + 524288;
    const long oCx64 = oBufB + 1048576, oF64 = oBufB + 3145728;
    const long oCx128 = oBufA, oXeFeat = oBufA, oXsFeat = oBufA;
    const long oXs128 = oBufB;                    // [2,150,128,128]

    // ================= PPM head =================
    { long tot = (long)B * 2048 * HW16;           // conv5 -> concat ch 0..2047
      resize_bilinear_kernel<<<CDIV(tot,256),256,0,stream>>>(
          conv5, 2048*HW16, 2048, 16,16, ws+oPpmCat, 4096*HW16, 16,16,
          nullptr, 0, tot); }
    const int scales[4] = {1, 2, 3, 6};
    for (int j = 0; j < 4; ++j) {
        int s = scales[j];
        { long tot = (long)B * 2048 * s * s;
          adaptive_pool_kernel<<<CDIV(tot,256),256,0,stream>>>(
              conv5, 2048, 16, 16, s, ws+oPool, tot); }
        { long tot = (long)B * 2048 * HW16;
          resize_bilinear_kernel<<<CDIV(tot,256),256,0,stream>>>(
              ws+oPool, (long)2048*s*s, 2048, s, s,
              ws+oPoolRes, 2048*HW16, 16, 16, nullptr, 0, tot); }
        { dim3 g(HW16/128, CDIV(512,64), B);
          conv_wmma_kernel<1,1><<<g,256,0,stream>>>(
              ws+oPoolRes, 2048*HW16, 2048, 16,16,
              ppm_w[j], ppm_s[j], ppm_b[j],
              ws+oPpmCat + (2048 + 512*j)*HW16, 4096*HW16, 512); }
    }
    { dim3 g(HW16/128, CDIV(256,64), B);          // ppm_last 3x3 4096->256
      conv_wmma_kernel<3,1><<<g,256,0,stream>>>(
          ws+oPpmCat, 4096*HW16, 4096, 16,16,
          ppl_w, ppl_s, ppl_b, ws+oF16, 256*HW16, 256); }

    // ================= FPN top-down =================
    // i = 2 (32^2)
    { dim3 g(HW32/128, CDIV(256,64), B);
      conv_wmma_kernel<1,1><<<g,256,0,stream>>>(
          conv2, 1024*HW32, 1024, 32,32, fin_w[2], fin_s[2], fin_b[2],
          ws+oCx32, 256*HW32, 256); }
    { long tot = (long)B * 256 * HW32;
      resize_bilinear_kernel<<<CDIV(tot,256),256,0,stream>>>(
          ws+oF16, 256*HW16, 256, 16,16, ws+oF32, 256*HW32, 32,32,
          ws+oCx32, 256*HW32, tot); }
    { dim3 g(HW32/128, CDIV(256,64), B);
      conv_wmma_kernel<3,1><<<g,256,0,stream>>>(
          ws+oF32, 256*HW32, 256, 32,32, fout_w[2], fout_s[2], fout_b[2],
          ws+oOut32, 256*HW32, 256); }
    // i = 1 (64^2)
    { dim3 g(HW64/128, CDIV(256,64), B);
      conv_wmma_kernel<1,1><<<g,256,0,stream>>>(
          conv1, 512*HW64, 512, 64,64, fin_w[1], fin_s[1], fin_b[1],
          ws+oCx64, 256*HW64, 256); }
    { long tot = (long)B * 256 * HW64;
      resize_bilinear_kernel<<<CDIV(tot,256),256,0,stream>>>(
          ws+oF32, 256*HW32, 256, 32,32, ws+oF64, 256*HW64, 64,64,
          ws+oCx64, 256*HW64, tot); }
    { dim3 g(HW64/128, CDIV(256,64), B);
      conv_wmma_kernel<3,1><<<g,256,0,stream>>>(
          ws+oF64, 256*HW64, 256, 64,64, fout_w[1], fout_s[1], fout_b[1],
          ws+oOut64, 256*HW64, 256); }
    // i = 0 (128^2)
    { dim3 g(HW128/128, CDIV(256,64), B);
      conv_wmma_kernel<1,1><<<g,256,0,stream>>>(
          conv0, 256*HW128, 256, 128,128, fin_w[0], fin_s[0], fin_b[0],
          ws+oCx128, 256*HW128, 256); }
    { long tot = (long)B * 256 * HW128;
      resize_bilinear_kernel<<<CDIV(tot,256),256,0,stream>>>(
          ws+oF64, 256*HW64, 256, 64,64, ws+oF128, 256*HW128, 128,128,
          ws+oCx128, 256*HW128, tot); }
    { dim3 g(HW128/128, CDIV(256,64), B);         // -> fusion channels 0..255
      conv_wmma_kernel<3,1><<<g,256,0,stream>>>(
          ws+oF128, 256*HW128, 256, 128,128, fout_w[0], fout_s[0], fout_b[0],
          ws+oFusion, 1024*HW128, 256); }
    // fusion channels 256.. / 512.. / 768..
    { long tot = (long)B * 256 * HW128;
      resize_bilinear_kernel<<<CDIV(tot,256),256,0,stream>>>(
          ws+oOut64, 256*HW64, 256, 64,64, ws+oFusion + 256*HW128,
          1024*HW128, 128,128, nullptr, 0, tot);
      resize_bilinear_kernel<<<CDIV(tot,256),256,0,stream>>>(
          ws+oOut32, 256*HW32, 256, 32,32, ws+oFusion + 512*HW128,
          1024*HW128, 128,128, nullptr, 0, tot);
      resize_bilinear_kernel<<<CDIV(tot,256),256,0,stream>>>(
          ws+oF16, 256*HW16, 256, 16,16, ws+oFusion + 768*HW128,
          1024*HW128, 128,128, nullptr, 0, tot); }

    // ================= euclidean logits =================
    { dim3 g(HW128/128, CDIV(256,64), B);
      conv_wmma_kernel<3,1><<<g,256,0,stream>>>(
          ws+oFusion, 1024*HW128, 1024, 128,128, lc_w, lc_s, lc_b,
          ws+oXeFeat, 256*HW128, 256); }
    { dim3 g(HW128/128, CDIV(150,64), B);         // 1x1 256->150 + bias
      conv_wmma_kernel<1,2><<<g,256,0,stream>>>(
          ws+oXeFeat, 256*HW128, 256, 128,128, last_w, nullptr, last_b,
          ws+oBufC, 150*HW128, 150); }

    // ================= sphere branch =================
    { dim3 g(HW128/128, CDIV(256,64), B);
      conv_wmma_kernel<3,1><<<g,256,0,stream>>>(
          ws+oFusion, 1024*HW128, 1024, 128,128, sp_w, sp_s, sp_b,
          ws+oXsFeat, 256*HW128, 256); }
    { long tot = (long)B * 256 * HW32;
      maxpool4_kernel<<<CDIV(tot,256),256,0,stream>>>(
          ws+oXsFeat, 256, 128, 128, ws+oMp, tot); }
    normalize_fc_kernel<<<1,256,0,stream>>>(fcw, ws+oWnT, 256, 150);
    { dim3 g(HW32/128, CDIV(150,64), B);          // mp @ wn : raw matmul
      conv_wmma_kernel<1,0><<<g,256,0,stream>>>(
          ws+oMp, 256*HW32, 256, 32,32, ws+oWnT, nullptr, nullptr,
          ws+oXs32, 150*HW32, 150); }
    { long tot = (long)B * 150 * HW128;           // logits 32^2 -> 128^2
      resize_bilinear_kernel<<<CDIV(tot,256),256,0,stream>>>(
          ws+oXs32, 150*HW32, 150, 32,32, ws+oXs128, 150*HW128, 128,128,
          nullptr, 0, tot); }

    // ================= scores + outputs =================
    { long tot = (long)B * HW128;
      kl_score_kernel<<<CDIV(tot,256),256,0,stream>>>(
          ws+oBufC, ws+oXs128, ws+oScore, 150, (int)HW128, tot); }
    { long tot = (long)B * HW512;                 // fused resize+softmax
      resize_softmax_kernel<<<CDIV(tot,256),256,0,stream>>>(
          ws+oBufC, 150, 128, 128, out, 512, 512, tot); }
    { long tot = (long)B * HW512;                 // scores -> 512^2
      resize_bilinear_kernel<<<CDIV(tot,256),256,0,stream>>>(
          ws+oScore, HW128, 1, 128, 128, out + (long)B * 150 * HW512,
          HW512, 512, 512, nullptr, 0, tot); }
}
